// SwinTransformerBlockFoldBack_38972533244326
// MI455X (gfx1250) — matrix operand order repaired and checked by reading, not compiled, not run on hardware
//
#include <hip/hip_runtime.h>
#include <math.h>
#include <stdint.h>

typedef __attribute__((ext_vector_type(16))) _Float16 v16h;
typedef __attribute__((ext_vector_type(8)))  float    v8f;

static constexpr int Bsz = 32;
static constexpr int Hh  = 56;
static constexpr int Ww  = 56;
static constexpr int C   = 192;
static constexpr int T   = Bsz * Hh * Ww;   // 100352 tokens
static constexpr int NHd = 6;
static constexpr int HD  = 32;              // head dim
static constexpr int WSz = 7;
static constexpr int NWIN = (Hh / WSz) * (Ww / WSz); // 64 windows / image
static constexpr int HID = 384;

__device__ __forceinline__ float gelu_exact(float x) {
  return 0.5f * x * (1.0f + erff(x * 0.70710678118654752f));
}

__device__ __forceinline__ v8f wmma16x16x32(v16h a, v16h b, v8f c) {
  // D = A(16x32 f16) * B(32x16 f16) + C(16x16 f32)
  return __builtin_amdgcn_wmma_f32_16x16x32_f16(false, a, false, b, (short)0, c,
                                                false, false);
}

// Async global -> LDS copies (CDNA5 TDM-lite path, tracked by ASYNCcnt).
// ldsoff = wave-relative LDS byte address (low 32 bits of a generic pointer
// to a __shared__ object), gaddr = 64-bit global address.
__device__ __forceinline__ void async_copy_b64(uint32_t ldsoff, uint64_t gaddr) {
  asm volatile("global_load_async_to_lds_b64 %0, %1, off"
               :: "v"(ldsoff), "v"(gaddr) : "memory");
}
__device__ __forceinline__ void async_copy_b128(uint32_t ldsoff, uint64_t gaddr) {
  asm volatile("global_load_async_to_lds_b128 %0, %1, off"
               :: "v"(ldsoff), "v"(gaddr) : "memory");
}
__device__ __forceinline__ void wait_async0() {
  asm volatile("s_wait_asynccnt 0" ::: "memory");
}
__device__ __forceinline__ uint32_t lds_off(const void* p) {
  return (uint32_t)(uintptr_t)p;
}

// ---------------------------------------------------------------------------
// Generic tiled WMMA GEMM:  out[M,N] = act( A[M,K] @ W[N,K]^T + bias ) (+res)
// Block = 256 threads (8 waves). Tile: 64(M) x 64(N), K-step 32.
// Wave w: M-subtile (w&3), N-group (w>>2) covering two 16-col subtiles that
// share one A fragment (2 accumulators / wave).
// Requires M%64==0, N%64==0, K%32==0 (true for every call site here).
// ---------------------------------------------------------------------------
__global__ __launch_bounds__(256)
void gemm_wmma_kernel(const _Float16* __restrict__ A,
                      const _Float16* __restrict__ W,
                      const float* __restrict__ bias,
                      const float* __restrict__ residual,
                      void* __restrict__ out,
                      int M, int N, int K,
                      int out_f16, int accumulate, int do_gelu)
{
  __shared__ _Float16 As[64][40];   // 64 x 32 (+pad)
  __shared__ _Float16 Bs[64][40];   // 64 x 32 (+pad)

  const int tid  = threadIdx.x;
  const int lane = tid & 31;
  const int wv   = tid >> 5;     // 0..7
  const int mt   = wv & 3;       // M sub-tile (16 rows each)
  const int ng   = wv >> 2;      // N group (32 cols each)
  const int r    = lane & 15;
  const int hi   = lane >> 4;    // lane group
  const int m0   = blockIdx.x * 64;
  const int n0   = blockIdx.y * 64;

  v8f acc0 = {}, acc1 = {};

  for (int kk = 0; kk < K; kk += 32) {
    // stage A (64x32) and B (64x32) tiles: async global->LDS, b64 per slot
    for (int idx = tid; idx < 512; idx += 256) {
      int row = idx >> 3, cs = (idx & 7) * 4;
      async_copy_b64(lds_off(&As[row][cs]),
                     (uint64_t)(uintptr_t)&A[(size_t)(m0 + row) * K + kk + cs]);
      async_copy_b64(lds_off(&Bs[row][cs]),
                     (uint64_t)(uintptr_t)&W[(size_t)(n0 + row) * K + kk + cs]);
    }
    if (kk + 32 < K)  // emits global_prefetch_b8 for next A tile
      __builtin_prefetch(&A[(size_t)(m0 + (tid >> 3)) * K + kk + 32], 0, 0);
    wait_async0();
    __syncthreads();

    // A fragment: row = lane%16; lanes<16 hold K 0..15, lanes>=16 hold K 16..31
    v16h a, b0, b1;
    const int arow = mt * 16 + r;
    const int k0 = hi * 16;
#pragma unroll
    for (int j = 0; j < 16; j++) {
      a[j]     = As[arow][k0 + j];
    }
    // B fragments: col = lane%16; lanes<16 hold K 0..15, lanes>=16 hold K 16..31
    const int bcol = ng * 32 + r;
    const int kb = hi * 16;
#pragma unroll
    for (int j = 0; j < 16; j++) { b0[j] = Bs[bcol][kb + j]; b1[j] = Bs[bcol + 16][kb + j]; }

    acc0 = wmma16x16x32(a, b0, acc0);
    acc1 = wmma16x16x32(a, b1, acc1);
    __syncthreads();
  }

  // C/D: vgpr i -> row i (+8 for lanes>=16), col = lane%16
  const int col0 = n0 + ng * 32 + r;
  const float bv0 = bias ? bias[col0] : 0.0f;
  const float bv1 = bias ? bias[col0 + 16] : 0.0f;
#pragma unroll
  for (int i = 0; i < 8; i++) {
    int row = m0 + mt * 16 + hi * 8 + i;
    float v0 = acc0[i] + bv0;
    float v1 = acc1[i] + bv1;
    if (do_gelu) { v0 = gelu_exact(v0); v1 = gelu_exact(v1); }
    size_t oi0 = (size_t)row * N + col0;
    size_t oi1 = oi0 + 16;
    if (residual) { v0 += residual[oi0]; v1 += residual[oi1]; }
    if (out_f16) {
      ((_Float16*)out)[oi0] = (_Float16)v0;
      ((_Float16*)out)[oi1] = (_Float16)v1;
    } else if (accumulate) {
      ((float*)out)[oi0] += v0;
      ((float*)out)[oi1] += v1;
    } else {
      ((float*)out)[oi0] = v0;
      ((float*)out)[oi1] = v1;
    }
  }
}

// ---------------------------------------------------------------------------
// Window attention: one block (64 threads = 2 waves) per (window, head).
// 49 tokens padded to 64; S = (q*scale) k^T via WMMA, softmax, P v via WMMA.
// K/V tiles staged via async global->LDS (b128); q scaled through VGPRs.
// ---------------------------------------------------------------------------
__global__ __launch_bounds__(64)
void window_attn_kernel(const _Float16* __restrict__ qkv,   // [T, 3C] f16
                        _Float16* __restrict__ attnout)     // [T, C]  f16
{
  __shared__ _Float16 qs[64][40];
  __shared__ _Float16 ks[64][40];
  __shared__ _Float16 vs[64][40];
  __shared__ float    Sm[64][65];
  __shared__ _Float16 Pm[64][72];

  const int t    = threadIdx.x;
  const int head = blockIdx.x % NHd;
  const int wi   = blockIdx.x / NHd;
  const int b    = wi >> 6;
  const int wrow = (wi >> 3) & 7;
  const int wcol = wi & 7;

  // zero-pad rows 49..63 (DS stores); async writes only touch rows < 49,
  // so DS and ASYNC LDS traffic never overlap (the counters are unordered).
  if (t >= 49) {
    for (int j = 0; j < 32; j++) {
      qs[t][j] = (_Float16)0.f; ks[t][j] = (_Float16)0.f; vs[t][j] = (_Float16)0.f;
    }
  }
  for (int j = 0; j < 64; j++) Pm[t][j] = (_Float16)0.f;

  if (t < 49) {
    int hh = wrow * WSz + t / WSz;
    int ww = wcol * WSz + t % WSz;
    size_t tok = (size_t)b * (Hh * Ww) + (size_t)hh * Ww + ww;
    const _Float16* qp = qkv + tok * (3 * C) + head * HD;
    const _Float16* kp = qp + C;
    const _Float16* vp = qp + 2 * C;
#pragma unroll
    for (int d = 0; d < HD; d += 8) {   // 16B chunks, 16B-aligned
      async_copy_b128(lds_off(&ks[t][d]), (uint64_t)(uintptr_t)&kp[d]);
      async_copy_b128(lds_off(&vs[t][d]), (uint64_t)(uintptr_t)&vp[d]);
    }
    const float scale = 0.17677669529663689f;   // 1/sqrt(32)
    for (int d = 0; d < HD; d++)
      qs[t][d] = (_Float16)((float)qp[d] * scale);
  }
  wait_async0();
  __syncthreads();

  const int wv = t >> 5, lane = t & 31;
  const int r = lane & 15, hi = lane >> 4;
  const int k0 = hi * 8, kb = hi * 16;

  // S = q k^T  (64x64, K=32)
  for (int mi = 0; mi < 2; mi++) {
    int mtl = wv * 2 + mi;
    v16h a;
#pragma unroll
    for (int j = 0; j < 16; j++) {
      a[j]     = qs[mtl * 16 + r][kb + j];
    }
    for (int ntl = 0; ntl < 4; ntl++) {
      v16h bf;
#pragma unroll
      for (int j = 0; j < 16; j++) bf[j] = ks[ntl * 16 + r][kb + j];
      v8f acc = {};
      acc = wmma16x16x32(a, bf, acc);
#pragma unroll
      for (int i = 0; i < 8; i++)
        Sm[mtl * 16 + hi * 8 + i][ntl * 16 + r] = acc[i];
    }
  }
  __syncthreads();

  // row softmax over the 49 valid columns
  if (t < 49) {
    float mx = -1e30f;
    for (int j = 0; j < 49; j++) mx = fmaxf(mx, Sm[t][j]);
    float s = 0.f;
    for (int j = 0; j < 49; j++) s += expf(Sm[t][j] - mx);
    float inv = 1.0f / s;
    for (int j = 0; j < 49; j++) Pm[t][j] = (_Float16)(expf(Sm[t][j] - mx) * inv);
  }
  __syncthreads();

  // out = P v   (64x32, K=64 in two 32-steps)
  for (int mi = 0; mi < 2; mi++) {
    int mtl = wv * 2 + mi;
    for (int ntl = 0; ntl < 2; ntl++) {
      v8f acc = {};
      for (int kk = 0; kk < 64; kk += 32) {
        v16h a, bf;
#pragma unroll
        for (int j = 0; j < 16; j++) {
          a[j]     = Pm[mtl * 16 + r][kk + kb + j];
        }
#pragma unroll
        for (int j = 0; j < 16; j++) bf[j] = vs[kk + kb + j][ntl * 16 + r];
        acc = wmma16x16x32(a, bf, acc);
      }
#pragma unroll
      for (int i = 0; i < 8; i++) {
        int m = mtl * 16 + hi * 8 + i;
        if (m < 49) {
          int hh = wrow * WSz + m / WSz;
          int ww = wcol * WSz + m % WSz;
          size_t tok = (size_t)b * (Hh * Ww) + (size_t)hh * Ww + ww;
          attnout[tok * C + head * HD + ntl * 16 + r] = (_Float16)acc[i];
        }
      }
    }
  }
}

// ---------------------------------------------------------------------------
// LayerNorm over C=192, one block (64 threads) per token, f16 output.
// ---------------------------------------------------------------------------
__global__ __launch_bounds__(64)
void layernorm_f16_kernel(const float* __restrict__ x, const float* __restrict__ g,
                          const float* __restrict__ bt, _Float16* __restrict__ out)
{
  __shared__ float red[64];
  __shared__ float stat;
  const int t = threadIdx.x;
  const float* row = x + (size_t)blockIdx.x * C;
  float v0 = row[t], v1 = row[t + 64], v2 = row[t + 128];
  red[t] = v0 + v1 + v2;
  __syncthreads();
  if (t == 0) { float s = 0; for (int i = 0; i < 64; i++) s += red[i]; stat = s / C; }
  __syncthreads();
  float m = stat;
  float d0 = v0 - m, d1 = v1 - m, d2 = v2 - m;
  __syncthreads();
  red[t] = d0 * d0 + d1 * d1 + d2 * d2;
  __syncthreads();
  if (t == 0) { float s = 0; for (int i = 0; i < 64; i++) s += red[i];
                stat = rsqrtf(s / C + 1e-5f); }
  __syncthreads();
  float rs = stat;
  _Float16* o = out + (size_t)blockIdx.x * C;
  o[t]       = (_Float16)(d0 * rs * g[t]       + bt[t]);
  o[t + 64]  = (_Float16)(d1 * rs * g[t + 64]  + bt[t + 64]);
  o[t + 128] = (_Float16)(d2 * rs * g[t + 128] + bt[t + 128]);
}

// depthwise 3x3 (SAME, groups=C) + exact GELU, NHWC
__global__ void dwconv_gelu_kernel(const _Float16* __restrict__ xn,
                                   const float* __restrict__ dw,
                                   _Float16* __restrict__ out)
{
  size_t idx = (size_t)blockIdx.x * blockDim.x + threadIdx.x;
  if (idx >= (size_t)T * C) return;
  int c = (int)(idx % C);
  size_t tk = idx / C;
  int w = (int)(tk % Ww);
  int h = (int)((tk / Ww) % Hh);
  int b = (int)(tk / ((size_t)Hh * Ww));
  float s = 0.f;
#pragma unroll
  for (int kh = 0; kh < 3; kh++)
#pragma unroll
    for (int kw = 0; kw < 3; kw++) {
      int ih = h + kh - 1, iw = w + kw - 1;
      if (ih >= 0 && ih < Hh && iw >= 0 && iw < Ww)
        s += (float)xn[(((size_t)b * Hh + ih) * Ww + iw) * C + c] *
             dw[c * 9 + kh * 3 + kw];
    }
  out[idx] = (_Float16)gelu_exact(s);
}

__global__ __launch_bounds__(192)
void gap_kernel(const _Float16* __restrict__ xn, float* __restrict__ gap)
{
  int b = blockIdx.x, c = threadIdx.x;
  float s = 0.f;
  for (int i = 0; i < Hh * Ww; i++)
    s += (float)xn[((size_t)b * (Hh * Ww) + i) * C + c];
  gap[b * C + c] = s / (float)(Hh * Ww);
}

__global__ __launch_bounds__(192)
void gate_kernel(const float* __restrict__ gap, const float* __restrict__ gw,
                 const float* __restrict__ gb, float* __restrict__ gate)
{
  int b = blockIdx.x, o = threadIdx.x;
  float s = gb[o];
  for (int cc = 0; cc < C; cc++) s += gap[b * C + cc] * gw[o * C + cc];
  gate[b * C + o] = 1.0f / (1.0f + expf(-s));
}

__global__ void addgate_kernel(float* __restrict__ xf, const _Float16* __restrict__ xn,
                               const float* __restrict__ gate)
{
  size_t idx = (size_t)blockIdx.x * blockDim.x + threadIdx.x;
  if (idx >= (size_t)T * C) return;
  int c = (int)(idx % C);
  int b = (int)((idx / C) / ((size_t)Hh * Ww));
  xf[idx] += (float)xn[idx] * gate[b * C + c];
}

__global__ __launch_bounds__(192)
void mean_over_h_kernel(const float* __restrict__ xf, float* __restrict__ mt)
{
  int b = blockIdx.x / Ww, w = blockIdx.x % Ww, c = threadIdx.x;
  float s = 0.f;
  for (int h = 0; h < Hh; h++)
    s += xf[(((size_t)b * Hh + h) * Ww + w) * C + c];
  mt[((size_t)b * C + c) * Ww + w] = s / (float)Hh;
}

__global__ __launch_bounds__(192)
void mean_over_w_kernel(const float* __restrict__ xf, float* __restrict__ ms)
{
  int b = blockIdx.x / Hh, h = blockIdx.x % Hh, c = threadIdx.x;
  float s = 0.f;
  for (int w = 0; w < Ww; w++)
    s += xf[(((size_t)b * Hh + h) * Ww + w) * C + c];
  ms[((size_t)b * C + c) * Hh + h] = s / (float)Ww;
}

__global__ __launch_bounds__(192)
void axis_logits_kernel(const float* __restrict__ m, const float* __restrict__ wgt,
                        const float* __restrict__ bias, float* __restrict__ outl, int Len)
{
  int b = blockIdx.x / Len, p = blockIdx.x % Len, o = threadIdx.x;
  float s = bias[o];
  for (int cc = 0; cc < C; cc++)
    s += wgt[o * C + cc] * m[((size_t)b * C + cc) * Len + p];
  outl[((size_t)b * C + o) * Len + p] = s;
}

__global__ __launch_bounds__(64)
void softmax_lastdim_kernel(float* __restrict__ data, int n)
{
  float* row = data + (size_t)blockIdx.x * n;
  __shared__ float red[64];
  __shared__ float bc;
  int t = threadIdx.x;
  float v = (t < n) ? row[t] : -1e30f;
  red[t] = v;
  __syncthreads();
  if (t == 0) { float m = -1e30f; for (int i = 0; i < 64; i++) m = fmaxf(m, red[i]); bc = m; }
  __syncthreads();
  float m = bc;
  float e = (t < n) ? expf(v - m) : 0.f;
  __syncthreads();
  red[t] = e;
  __syncthreads();
  if (t == 0) { float s = 0; for (int i = 0; i < 64; i++) s += red[i]; bc = s; }
  __syncthreads();
  if (t < n) row[t] = e / bc;
}

__global__ void scale_xf_kernel(const float* __restrict__ xf, const float* __restrict__ a_s,
                                const float* __restrict__ a_t, _Float16* __restrict__ out)
{
  size_t idx = (size_t)blockIdx.x * blockDim.x + threadIdx.x;
  if (idx >= (size_t)T * C) return;
  int c = (int)(idx % C);
  size_t tk = idx / C;
  int w = (int)(tk % Ww);
  int h = (int)((tk / Ww) % Hh);
  int b = (int)(tk / ((size_t)Hh * Ww));
  float v = xf[idx] * a_s[((size_t)b * C + c) * Hh + h] *
            a_t[((size_t)b * C + c) * Ww + w] * 2.0f;
  out[idx] = (_Float16)v;
}

__global__ void f32_to_f16_kernel(const float* __restrict__ src,
                                  _Float16* __restrict__ dst, int n)
{
  int i = blockIdx.x * blockDim.x + threadIdx.x;
  if (i < n) dst[i] = (_Float16)src[i];
}

// ---------------------------------------------------------------------------
extern "C" void kernel_launch(void* const* d_in, const int* in_sizes, int n_in,
                              void* d_out, int out_size, void* d_ws, size_t ws_size,
                              hipStream_t stream)
{
  (void)in_sizes; (void)n_in; (void)out_size; (void)ws_size;

  const float* x      = (const float*)d_in[0];
  const float* ln1_g  = (const float*)d_in[3];
  const float* ln1_b  = (const float*)d_in[4];
  const float* qkv_w  = (const float*)d_in[5];
  const float* qkv_b  = (const float*)d_in[6];
  const float* proj_w = (const float*)d_in[7];
  const float* proj_b = (const float*)d_in[8];
  const float* dw_w   = (const float*)d_in[9];
  const float* pw_w   = (const float*)d_in[10];
  const float* gate_w = (const float*)d_in[11];
  const float* gate_b = (const float*)d_in[12];
  const float* temp_w = (const float*)d_in[13];
  const float* temp_b = (const float*)d_in[14];
  const float* spec_w = (const float*)d_in[15];
  const float* spec_b = (const float*)d_in[16];
  const float* fuse_w = (const float*)d_in[17];
  const float* ln2_g  = (const float*)d_in[18];
  const float* ln2_b  = (const float*)d_in[19];
  const float* mlp1_w = (const float*)d_in[20];
  const float* mlp1_b = (const float*)d_in[21];
  const float* mlp2_w = (const float*)d_in[22];
  const float* mlp2_b = (const float*)d_in[23];

  char* base = (char*)d_ws;
  size_t off = 0;
  auto alloc = [&](size_t bytes) {
    void* p = base + off;
    off += (bytes + 255) & ~(size_t)255;
    return p;
  };

  _Float16* wqkv_h  = (_Float16*)alloc((size_t)3 * C * C * 2);
  _Float16* wproj_h = (_Float16*)alloc((size_t)C * C * 2);
  _Float16* wpw_h   = (_Float16*)alloc((size_t)C * C * 2);
  _Float16* wfuse_h = (_Float16*)alloc((size_t)C * C * 2);
  _Float16* wm1_h   = (_Float16*)alloc((size_t)HID * C * 2);
  _Float16* wm2_h   = (_Float16*)alloc((size_t)C * HID * 2);
  _Float16* xn_h    = (_Float16*)alloc((size_t)T * C * 2);
  _Float16* qkv_h   = (_Float16*)alloc((size_t)T * 3 * C * 2);
  _Float16* attn_h  = (_Float16*)alloc((size_t)T * C * 2);
  _Float16* ld_h    = (_Float16*)alloc((size_t)T * C * 2);
  float*    xf      = (float*)   alloc((size_t)T * C * 4);
  float*    gapb    = (float*)   alloc((size_t)Bsz * C * 4);
  float*    gateb   = (float*)   alloc((size_t)Bsz * C * 4);
  float*    m_t     = (float*)   alloc((size_t)Bsz * C * Ww * 4);
  float*    m_s     = (float*)   alloc((size_t)Bsz * C * Hh * 4);
  float*    a_t     = (float*)   alloc((size_t)Bsz * C * Ww * 4);
  float*    a_s     = (float*)   alloc((size_t)Bsz * C * Hh * 4);
  _Float16* hidden  = qkv_h;   // reuse: QKV buffer is dead after attention
  _Float16* ln2_h   = xn_h;    // reuse: xn dead after gate/dwconv consumers
  _Float16* xf2_h   = ld_h;    // reuse: dwconv output dead after pw GEMM

  auto cvt = [&](const float* s, _Float16* d, int n) {
    f32_to_f16_kernel<<<(n + 255) / 256, 256, 0, stream>>>(s, d, n);
  };
  cvt(qkv_w, wqkv_h, 3 * C * C);
  cvt(proj_w, wproj_h, C * C);
  cvt(pw_w, wpw_h, C * C);
  cvt(fuse_w, wfuse_h, C * C);
  cvt(mlp1_w, wm1_h, HID * C);
  cvt(mlp2_w, wm2_h, C * HID);

  // LN1 -> xn (f16), global-average gate
  layernorm_f16_kernel<<<T, 64, 0, stream>>>(x, ln1_g, ln1_b, xn_h);
  gap_kernel<<<Bsz, C, 0, stream>>>(xn_h, gapb);
  gate_kernel<<<Bsz, C, 0, stream>>>(gapb, gate_w, gate_b, gateb);

  // QKV projection (WMMA)
  dim3 gQKV(T / 64, (3 * C) / 64);
  gemm_wmma_kernel<<<gQKV, 256, 0, stream>>>(xn_h, wqkv_h, qkv_b, nullptr,
                                             qkv_h, T, 3 * C, C, 1, 0, 0);

  // window attention (WMMA, one block per window-head)
  window_attn_kernel<<<Bsz * NWIN * NHd, 64, 0, stream>>>(qkv_h, attn_h);

  // proj -> xf (f32, init)
  dim3 gP(T / 64, C / 64);
  gemm_wmma_kernel<<<gP, 256, 0, stream>>>(attn_h, wproj_h, proj_b, nullptr,
                                           xf, T, C, C, 0, 0, 0);

  // local conv branch: dw3x3+GELU then pointwise (accumulate into xf)
  dwconv_gelu_kernel<<<(T * C + 255) / 256, 256, 0, stream>>>(xn_h, dw_w, ld_h);
  gemm_wmma_kernel<<<gP, 256, 0, stream>>>(ld_h, wpw_h, nullptr, nullptr,
                                           xf, T, C, C, 0, 1, 0);
  addgate_kernel<<<(T * C + 255) / 256, 256, 0, stream>>>(xf, xn_h, gateb);

  // spectral-temporal attention
  mean_over_h_kernel<<<Bsz * Ww, C, 0, stream>>>(xf, m_t);
  mean_over_w_kernel<<<Bsz * Hh, C, 0, stream>>>(xf, m_s);
  axis_logits_kernel<<<Bsz * Ww, C, 0, stream>>>(m_t, temp_w, temp_b, a_t, Ww);
  axis_logits_kernel<<<Bsz * Hh, C, 0, stream>>>(m_s, spec_w, spec_b, a_s, Hh);
  softmax_lastdim_kernel<<<Bsz * C, 64, 0, stream>>>(a_t, Ww);
  softmax_lastdim_kernel<<<Bsz * C, 64, 0, stream>>>(a_s, Hh);
  scale_xf_kernel<<<(T * C + 255) / 256, 256, 0, stream>>>(xf, a_s, a_t, xf2_h);

  // fuse 1x1 + residual (x) -> d_out (f32)
  gemm_wmma_kernel<<<gP, 256, 0, stream>>>(xf2_h, wfuse_h, nullptr, x,
                                           (float*)d_out, T, C, C, 0, 0, 0);

  // MLP with pre-norm residual
  layernorm_f16_kernel<<<T, 64, 0, stream>>>((const float*)d_out, ln2_g, ln2_b, ln2_h);
  dim3 gM1(T / 64, HID / 64);
  gemm_wmma_kernel<<<gM1, 256, 0, stream>>>(ln2_h, wm1_h, mlp1_b, nullptr,
                                            hidden, T, HID, C, 1, 0, 1);
  gemm_wmma_kernel<<<gP, 256, 0, stream>>>(hidden, wm2_h, mlp2_b, nullptr,
                                           (float*)d_out, T, C, HID, 0, 1, 0);
}
